// ExpertMoEClass_40450001994300
// MI455X (gfx1250) — matrix operand order repaired and checked by reading, not compiled, not run on hardware
//
#include <hip/hip_runtime.h>

// Problem constants (match reference)
#define T_TOK 1024
#define H_DIM 1024
#define I_DIM 2048
#define N_EXP 8
#define TOPK  2

#define TILE_M 16

typedef __attribute__((ext_vector_type(16))) __bf16 v16bf;
typedef __attribute__((ext_vector_type(8)))  float  v8f;

__device__ __forceinline__ __bf16 f2bf(float f) {
  unsigned int u = __builtin_bit_cast(unsigned int, f);
  unsigned int r = (u + 0x7FFFu + ((u >> 16) & 1u)) >> 16;   // round-to-nearest-even
  unsigned short s = (unsigned short)r;
  return __builtin_bit_cast(__bf16, s);
}

// ---------------------------------------------------------------------------
// Kernel 0: zero the output accumulator (we atomically accumulate into it)
// ---------------------------------------------------------------------------
__global__ void moe_zero_out(float* __restrict__ out, int n) {
  int i = blockIdx.x * blockDim.x + threadIdx.x;
  if (i < n) out[i] = 0.0f;
}

// ---------------------------------------------------------------------------
// Kernel 1: dense combine weights [T,E] = scatter(normalized top-k affinity)
// ---------------------------------------------------------------------------
__global__ void moe_combine(const float* __restrict__ aff,        // [T,E]
                            const long long* __restrict__ idx,    // [T,TOPK] i64
                            float* __restrict__ combine) {        // [T,E]
  int t = blockIdx.x * blockDim.x + threadIdx.x;
  if (t >= T_TOK) return;
  float c[N_EXP];
#pragma unroll
  for (int e = 0; e < N_EXP; ++e) c[e] = 0.0f;
  int e0 = (int)idx[t * TOPK + 0];
  int e1 = (int)idx[t * TOPK + 1];
  float a0 = aff[t * N_EXP + e0];
  float a1 = aff[t * N_EXP + e1];
  float inv = 1.0f / (a0 + a1);
  c[e0] += a0 * inv;           // += handles duplicate indices like the reference
  c[e1] += a1 * inv;
#pragma unroll
  for (int e = 0; e < N_EXP; ++e) combine[t * N_EXP + e] = c[e];
}

// ---------------------------------------------------------------------------
// Kernel 2a: hidden fp32 -> bf16, swizzled into the WMMA A-operand layout.
// Layout: [tokenTile][kchunk(H/32)][lane(32)][j(16)] bf16, 32B per lane-frag.
// A map (16-bit 16x32): laneHi=(k>>3)&1, j=(k&7)+(k>=16?8:0), lane=m+16*laneHi
// ---------------------------------------------------------------------------
__global__ __launch_bounds__(256) void conv_hidden_swz(const float* __restrict__ src,
                                                       __bf16* __restrict__ dst) {
  int idx = blockIdx.x * blockDim.x + threadIdx.x;       // T*H = 1M elements
  int t = idx >> 10;
  int h = idx & (H_DIM - 1);
  int tile = t >> 4, m = t & 15;
  int kc = h >> 5, kl = h & 31;
  int laneHi = (kl >> 3) & 1;
  int j = (kl & 7) + (kl >= 16 ? 8 : 0);
  int lane = m + (laneHi << 4);
  size_t off = ((size_t)(tile * (H_DIM / 32) + kc) * 32 + lane) * 16 + j;
  dst[off] = f2bf(src[idx]);
}

// ---------------------------------------------------------------------------
// Kernel 2b: weights fp32 -> bf16, swizzled into the WMMA B-operand layout.
// Source layout [E,K,N]; per 32(K)x16(N) tile: lane=(n&15)+16*((k>>3)&1),
// j=(k&7)+((k&31)>=16?8:0); tile = ((e*K/32 + k/32)*N/16 + n/16), 512 bf16 (1 KB) per tile.
// ---------------------------------------------------------------------------
template <int KDIM, int NDIM>
__global__ __launch_bounds__(256) void conv_weight_swz(const float* __restrict__ src,
                                                       __bf16* __restrict__ dst) {
  size_t idx = (size_t)blockIdx.x * blockDim.x + threadIdx.x;   // E*K*N elements
  int n = (int)(idx % NDIM);
  int k = (int)((idx / NDIM) % KDIM);
  int e = (int)(idx / ((size_t)KDIM * NDIM));
  int kc = k >> 5, kl = k & 31;
  int lane = (n & 15) + (((kl >> 3) & 1) << 4);
  int j = (kl & 7) + (kl >= 16 ? 8 : 0);
  size_t off = ((size_t)((e * (KDIM / 32) + kc) * (NDIM / 16)) + (n >> 4)) * 512
             + (size_t)lane * 16 + j;
  dst[off] = f2bf(src[idx]);
}

// ---------------------------------------------------------------------------
// Kernel 3: fused GLU expert MLP + weighted combine.
// grid = (T/16 token tiles, E experts); 256 threads = 8 waves (wave32).
// All WMMA operands are contiguous 32B per-lane fragments: A from pre-swizzled
// global hidden (L2/WGP$-resident, reused 16x), B from pre-swizzled bf16
// weights (2x global_load_b128), stage-2 A from LDS (2x ds_load_b128).
// ---------------------------------------------------------------------------
__global__ __launch_bounds__(256) void moe_expert_fused(
    const __bf16* __restrict__ hB,   // [T/16][H/32][32][16] bf16 (ws)
    const __bf16* __restrict__ wgB,  // gate: tiled [E][H/32][I/16][512]
    const __bf16* __restrict__ wuB,  // up:   tiled [E][H/32][I/16][512]
    const __bf16* __restrict__ wdB,  // down: tiled [E][I/32][H/16][512]
    const float*  __restrict__ combine,  // [T,E]
    float* __restrict__ out)             // [T,H] atomic accumulate
{
  // act tile 16 x I in A-operand layout: [kchunk(I/32)][lane(32)][j(16)] bf16
  __shared__ __align__(32) __bf16 lds_act[(I_DIM / 32) * 32 * 16];   // 64 KB
  __shared__ float lds_cmb[TILE_M];

  const int tile = blockIdx.x;
  const int e    = blockIdx.y;
  const int t0   = tile * TILE_M;
  const int tid  = threadIdx.x;
  const int wave = tid >> 5;
  const int lane = tid & 31;
  const int laneHi = lane >> 4;
  const int laneN  = lane & 15;

  if (tid < TILE_M) lds_cmb[tid] = combine[(t0 + tid) * N_EXP + e];
  __syncthreads();

  const __bf16* aBase = hB + (size_t)tile * (H_DIM / 32) * 512;

  // ---------------- Stage 1: act = silu(x Wg) * (x Wu) --------------------
  // I/16 = 128 column tiles, 16 per wave.
  for (int it = 0; it < I_DIM / (16 * 8); ++it) {
    const int nt = wave + it * 8;          // column tile index
    v8f accg = {};
    v8f accu = {};
    for (int kc = 0; kc < H_DIM / 32; ++kc) {
      const v16bf a  = *(const v16bf*)(aBase + (size_t)kc * 512 + lane * 16);
      const size_t bo = ((size_t)((e * (H_DIM / 32) + kc) * (I_DIM / 16)) + nt) * 512
                      + (size_t)lane * 16;
      const v16bf bg = *(const v16bf*)(wgB + bo);
      const v16bf bu = *(const v16bf*)(wuB + bo);
      if (kc + 1 < H_DIM / 32) {
        __builtin_prefetch(wgB + bo + (size_t)(I_DIM / 16) * 512, 0, 0);
        __builtin_prefetch(wuB + bo + (size_t)(I_DIM / 16) * 512, 0, 0);
      }
      accg = __builtin_amdgcn_wmma_f32_16x16x32_bf16(false, a, false, bg,
                                                     (short)0, accg, false, false);
      accu = __builtin_amdgcn_wmma_f32_16x16x32_bf16(false, a, false, bu,
                                                     (short)0, accu, false, false);
    }
    // C/D: VGPR r, lane -> M = r + 8*laneHi, N = nt*16 + laneN.
    // Store into LDS in A-operand layout for stage 2 (K dimension = I index).
    const int n = nt * 16 + laneN;
    const int kc2 = n >> 5, kl2 = n & 31;
    const int laneHi2 = (kl2 >> 3) & 1;
    const int j2 = (kl2 & 7) + (kl2 >= 16 ? 8 : 0);
#pragma unroll
    for (int r = 0; r < 8; ++r) {
      const int m = r + (laneHi << 3);
      const float g = accg[r];
      const float u = accu[r];
      const float s = g / (1.0f + __expf(-g));   // silu
      lds_act[((size_t)kc2 * 32 + (m + (laneHi2 << 4))) * 16 + j2] = f2bf(s * u);
    }
  }
  __syncthreads();

  // ---------------- Stage 2: out += (act Wd) * combine --------------------
  // H/16 = 64 column tiles, 8 per wave.
  for (int it = 0; it < H_DIM / (16 * 8); ++it) {
    const int nt = wave + it * 8;
    v8f acc = {};
    for (int kc = 0; kc < I_DIM / 32; ++kc) {
      const v16bf a = *(const v16bf*)(lds_act + (size_t)kc * 512 + lane * 16);
      const size_t bo = ((size_t)((e * (I_DIM / 32) + kc) * (H_DIM / 16)) + nt) * 512
                      + (size_t)lane * 16;
      const v16bf b = *(const v16bf*)(wdB + bo);
      if (kc + 1 < I_DIM / 32) {
        __builtin_prefetch(wdB + bo + (size_t)(H_DIM / 16) * 512, 0, 0);
      }
      acc = __builtin_amdgcn_wmma_f32_16x16x32_bf16(false, a, false, b,
                                                    (short)0, acc, false, false);
    }
#pragma unroll
    for (int r = 0; r < 8; ++r) {
      const int m = r + (laneHi << 3);
      atomicAdd(&out[(size_t)(t0 + m) * H_DIM + nt * 16 + laneN],
                acc[r] * lds_cmb[m]);
    }
  }
}

// ---------------------------------------------------------------------------
// Launch
// ---------------------------------------------------------------------------
extern "C" void kernel_launch(void* const* d_in, const int* in_sizes, int n_in,
                              void* d_out, int out_size, void* d_ws, size_t ws_size,
                              hipStream_t stream) {
  const float*     hidden = (const float*)d_in[0];      // [T,H] f32
  const float*     aff    = (const float*)d_in[1];      // [T,E] f32
  const long long* idx    = (const long long*)d_in[2];  // [T,TOPK] i64
  const float*     wg     = (const float*)d_in[3];      // [E,H,I] f32
  const float*     wu     = (const float*)d_in[4];      // [E,H,I] f32
  const float*     wd     = (const float*)d_in[5];      // [E,I,H] f32
  float*           out    = (float*)d_out;              // [T,H] f32

  // Workspace layout (bytes)
  char* ws = (char*)d_ws;
  float*  combine = (float*)ws;                               // 32 KB
  __bf16* hB      = (__bf16*)(ws + (32 << 10));               // 2 MB
  __bf16* wgB     = (__bf16*)(ws + (32 << 10) + (2u << 20));  // 32 MB
  __bf16* wuB     = (__bf16*)(ws + (32 << 10) + (34u << 20)); // 32 MB
  __bf16* wdB     = (__bf16*)(ws + (32 << 10) + (66u << 20)); // 32 MB

  // 1) zero output accumulator
  {
    int n = T_TOK * H_DIM;
    moe_zero_out<<<(n + 255) / 256, 256, 0, stream>>>(out, n);
  }
  // 2) combine weights
  moe_combine<<<(T_TOK + 255) / 256, 256, 0, stream>>>(aff, idx, combine);
  // 3) fp32 -> bf16 swizzle passes
  conv_hidden_swz<<<(T_TOK * H_DIM) / 256, 256, 0, stream>>>(hidden, hB);
  {
    size_t nw = (size_t)N_EXP * H_DIM * I_DIM;
    conv_weight_swz<H_DIM, I_DIM><<<(unsigned)(nw / 256), 256, 0, stream>>>(wg, wgB);
    conv_weight_swz<H_DIM, I_DIM><<<(unsigned)(nw / 256), 256, 0, stream>>>(wu, wuB);
    conv_weight_swz<I_DIM, H_DIM><<<(unsigned)(nw / 256), 256, 0, stream>>>(wd, wdB);
  }
  // 4) fused expert GLU MLP + weighted combine
  dim3 grid(T_TOK / TILE_M, N_EXP);
  moe_expert_fused<<<grid, 256, 0, stream>>>(hB, wgB, wuB, wdB, combine, out);
}
